// FFF_66958540145282
// MI455X (gfx1250) — matrix operand order repaired and checked by reading, not compiled, not run on hardware
//
#include <hip/hip_runtime.h>
#include <hip/hip_bf16.h>

// FFF tree-MLP fused as: routing + 2 big bf16 WMMA GEMMs.
//  GEMM-A: (8192 x 1056) @ (1056 x 4096)  -> relu * mixture -> Hs (bf16)
//  GEMM-B: (8192 x 4128) @ (4128 x 1024)  -> out (fp32)
// Biases folded in via K-padding. ~137 GFLOP, compute-bound on MI455X.
// v_wmma_f32_16x16x32_bf16, fp32 accumulation, register double-buffered
// k-loop + WGP-scope global_prefetch (inline asm, scope 0) to stage
// L2 -> WGP$ ahead of the fragment loads.

typedef __bf16 bf16;
typedef bf16 v16bf __attribute__((ext_vector_type(16)));
typedef bf16 v8bf  __attribute__((ext_vector_type(8)));
typedef float v8f  __attribute__((ext_vector_type(8)));

#define DEPTH   4
#define IN_F    1024
#define LEAF_W  256
#define OUT_F   1024
#define BATCH   8192
#define NLEAF   16
#define NNODE   15

#define KA 1056           // 1024 + 32 pad (bias column at 1024)
#define NA 4096           // 16 leaves * 256
#define KB 4128           // 4096 + 32 pad (bias columns 4096..4111)

// ---------------- conversion kernels (fp32 -> padded bf16) ----------------

__global__ __launch_bounds__(256) void cvt_x(const float* __restrict__ x,
                                             bf16* __restrict__ xb) {
  long idx = (long)blockIdx.x * blockDim.x + threadIdx.x;
  if (idx >= (long)BATCH * KA) return;
  int b = (int)(idx / KA), k = (int)(idx % KA);
  float v = (k < IN_F) ? x[(long)b * IN_F + k] : (k == IN_F ? 1.0f : 0.0f);
  xb[idx] = (bf16)v;
}

// W1t[n][k] = w1s[l][k][h], n = l*256+h; col 1024 = b1s[n]
__global__ __launch_bounds__(256) void cvt_w1(const float* __restrict__ w1,
                                              const float* __restrict__ b1,
                                              bf16* __restrict__ w1t) {
  long idx = (long)blockIdx.x * blockDim.x + threadIdx.x;
  if (idx >= (long)NA * KA) return;
  int n = (int)(idx / KA), k = (int)(idx % KA);
  int l = n >> 8, h = n & 255;
  float v;
  if (k < IN_F)        v = w1[(long)l * IN_F * LEAF_W + (long)k * LEAF_W + h];
  else if (k == IN_F)  v = b1[l * LEAF_W + h];
  else                 v = 0.0f;
  w1t[idx] = (bf16)v;
}

// W2t[o][k] = w2s_flat[k*1024 + o] for k<4096; cols 4096+l = b2s[l][o]
__global__ __launch_bounds__(256) void cvt_w2(const float* __restrict__ w2,
                                              const float* __restrict__ b2,
                                              bf16* __restrict__ w2t) {
  long idx = (long)blockIdx.x * blockDim.x + threadIdx.x;
  if (idx >= (long)OUT_F * KB) return;
  int o = (int)(idx / KB), k = (int)(idx % KB);
  float v;
  if (k < NA)            v = w2[(long)k * OUT_F + o];
  else if (k < NA + 16)  v = b2[(k - NA) * OUT_F + o];
  else                   v = 0.0f;
  w2t[idx] = (bf16)v;
}

// ---------------- routing: mixture weights (one wave per row) -------------

__global__ __launch_bounds__(256) void routing(const float* __restrict__ x,
                                               const float* __restrict__ nw,
                                               const float* __restrict__ nb,
                                               float* __restrict__ mix,
                                               bf16* __restrict__ Hs) {
  __shared__ float gsh[8][16];
  const int lane = threadIdx.x & 31;
  const int wave = threadIdx.x >> 5;
  const int row  = blockIdx.x * 8 + wave;

  float p[NNODE];
#pragma unroll
  for (int n = 0; n < NNODE; ++n) p[n] = 0.0f;

  const float* xr = x + (long)row * IN_F;
  for (int t = lane; t < IN_F; t += 32) {
    float xv = xr[t];
#pragma unroll
    for (int n = 0; n < NNODE; ++n) p[n] += xv * nw[n * IN_F + t];
  }
#pragma unroll
  for (int off = 16; off > 0; off >>= 1) {
#pragma unroll
    for (int n = 0; n < NNODE; ++n) p[n] += __shfl_xor(p[n], off, 32);
  }
  if (lane == 0) {
#pragma unroll
    for (int n = 0; n < NNODE; ++n)
      gsh[wave][n] = 1.0f / (1.0f + __expf(-(p[n] + nb[n])));
  }
  __syncthreads();

  if (lane < 16) {
    int leaf = lane;
    float m = 1.0f;
#pragma unroll
    for (int d = 0; d < DEPTH; ++d) {
      int node = (1 << d) - 1 + (leaf >> (DEPTH - d));
      int bit  = (leaf >> (DEPTH - 1 - d)) & 1;
      float be = gsh[wave][node];
      m *= bit ? be : (1.0f - be);
    }
    mix[row * 16 + leaf] = m;
    Hs[(long)row * KB + NA + leaf] = (bf16)m;       // bias-activation column
  } else {
    Hs[(long)row * KB + NA + lane] = (bf16)0.0f;    // zero pad cols 4112..4127
  }
}

// ---------------- WMMA GEMM: C(M,N) = A(M,K) * B(N-major, N,K) ------------

union Frag { v16bf v; v8bf h[2]; };
struct Frags {
  Frag  a[2];
  v16bf b[4];
};

__device__ __forceinline__ void load_frags(Frags& f, const bf16* const* aB,
                                           const bf16* const* bB, int kk) {
#pragma unroll
  for (int i = 0; i < 2; ++i) {
    f.a[i].h[0] = *(const v8bf*)(aB[i] + kk);
    f.a[i].h[1] = *(const v8bf*)(aB[i] + kk + 8);
  }
#pragma unroll
  for (int j = 0; j < 4; ++j) f.b[j] = *(const v16bf*)(bB[j] + kk);
}

__device__ __forceinline__ void do_wmma(v8f acc[2][4], const Frags& f) {
#pragma unroll
  for (int i = 0; i < 2; ++i)
#pragma unroll
    for (int j = 0; j < 4; ++j)
      acc[i][j] = __builtin_amdgcn_wmma_f32_16x16x32_bf16(
          false, f.a[i].v, false, f.b[j], (short)0, acc[i][j], false, false);
}

// WGP-scope prefetch (scope field 0 = default/omitted): fills every cache
// level on a miss.  __builtin_prefetch can only reach SE/DEV/SYS scopes,
// which stop at GL2 -- useless here since the workspace already sits in the
// 192MB L2.  Inline asm reaches the scope the builtin can't.
__device__ __forceinline__ void wgp_prefetch(const void* p) {
  asm volatile("global_prefetch_b8 %0, off" :: "v"(p));
}

__device__ __forceinline__ void prefetch_k(const bf16* const* aB,
                                           const bf16* const* bB, int kk) {
#pragma unroll
  for (int i = 0; i < 2; ++i) wgp_prefetch(aB[i] + kk);
#pragma unroll
  for (int j = 0; j < 4; ++j) wgp_prefetch(bB[j] + kk);
}

// EPI==0: relu * mixture -> bf16 Hs (row stride KB)
// EPI==1: fp32 -> out (row stride OUT_F)
template <int EPI>
__global__ __launch_bounds__(256) void gemm_wmma(const bf16* __restrict__ A,
                                                 const bf16* __restrict__ B,
                                                 int K,
                                                 float* __restrict__ outF,
                                                 bf16* __restrict__ outH,
                                                 const float* __restrict__ mix) {
  const int lane = threadIdx.x & 31;
  const int wave = threadIdx.x >> 5;
  const int wm   = wave & 3;               // 4 waves along M
  const int wn   = wave >> 2;              // 2 waves along N
  const int m0   = blockIdx.y * 128 + wm * 32;
  const int n0   = blockIdx.x * 128 + wn * 64;
  const int lm   = lane & 15;
  const int sel  = lane >> 4;

  const v8f zero = {0.f, 0.f, 0.f, 0.f, 0.f, 0.f, 0.f, 0.f};
  v8f acc[2][4];
#pragma unroll
  for (int i = 0; i < 2; ++i)
#pragma unroll
    for (int j = 0; j < 4; ++j) acc[i][j] = zero;

  // A fragment: lane m = lm, 16 contiguous K at sel*16
  const bf16* aB[2];
  aB[0] = A + (long)(m0 + lm) * K + sel * 16;
  aB[1] = A + (long)(m0 + 16 + lm) * K + sel * 16;
  // B fragment (N-major): lane n = lm, 16 contiguous K at sel*16
  const bf16* bB[4];
#pragma unroll
  for (int j = 0; j < 4; ++j)
    bB[j] = B + (long)(n0 + 16 * j + lm) * K + sel * 16;

  // Software-pipelined k-loop. Trip count K/32 is odd (33 / 129):
  // preload step 0, then pairwise double-buffered body, one peeled tail.
  Frags f0, f1;
  load_frags(f0, aB, bB, 0);
  for (int kk = 32; kk + 32 < K; kk += 64) {
    load_frags(f1, aB, bB, kk);
    prefetch_k(aB, bB, kk + 128);          // pull L2 -> WGP$ four steps ahead
    do_wmma(acc, f0);
    load_frags(f0, aB, bB, kk + 32);
    prefetch_k(aB, bB, kk + 160);
    do_wmma(acc, f1);
  }
  do_wmma(acc, f0);

  // C fragment layout: N = lm, M = sel*8 + r (per 16x16 tile)
  if (EPI == 0) {
    const int l = n0 >> 8;  // wave's 64-col span lies inside one leaf block
#pragma unroll
    for (int i = 0; i < 2; ++i) {
#pragma unroll
      for (int r = 0; r < 8; ++r) {
        const int M = m0 + 16 * i + sel * 8 + r;
        const float s = mix[M * 16 + l];
#pragma unroll
        for (int j = 0; j < 4; ++j) {
          const int N = n0 + 16 * j + lm;
          float v = acc[i][j][r];
          v = v > 0.0f ? v : 0.0f;
          outH[(long)M * KB + N] = (bf16)(v * s);
        }
      }
    }
  } else {
#pragma unroll
    for (int i = 0; i < 2; ++i)
#pragma unroll
      for (int j = 0; j < 4; ++j)
#pragma unroll
        for (int r = 0; r < 8; ++r) {
          const int M = m0 + 16 * i + sel * 8 + r;
          const int N = n0 + 16 * j + lm;
          outF[(long)M * OUT_F + N] = acc[i][j][r];
        }
  }
}

// ---------------------------------------------------------------------------

extern "C" void kernel_launch(void* const* d_in, const int* in_sizes, int n_in,
                              void* d_out, int out_size, void* d_ws, size_t ws_size,
                              hipStream_t stream) {
  (void)in_sizes; (void)n_in; (void)out_size; (void)ws_size;
  const float* x  = (const float*)d_in[0];
  const float* nw = (const float*)d_in[1];
  const float* nb = (const float*)d_in[2];
  const float* w1 = (const float*)d_in[3];
  const float* b1 = (const float*)d_in[4];
  const float* w2 = (const float*)d_in[5];
  const float* b2 = (const float*)d_in[6];
  float* out = (float*)d_out;

  char* ws = (char*)d_ws;
  size_t off = 0;
  bf16*  Xb  = (bf16*)(ws + off);  off += (size_t)BATCH * KA * 2;   // 16.5 MB
  bf16*  W1t = (bf16*)(ws + off);  off += (size_t)NA    * KA * 2;   //  8.3 MB
  bf16*  W2t = (bf16*)(ws + off);  off += (size_t)OUT_F * KB * 2;   //  8.1 MB
  float* mix = (float*)(ws + off); off += (size_t)BATCH * 16 * 4;   //  0.5 MB
  bf16*  Hs  = (bf16*)(ws + off);                                   // 64.5 MB

  {
    long n = (long)BATCH * KA;
    cvt_x<<<dim3((unsigned)((n + 255) / 256)), 256, 0, stream>>>(x, Xb);
  }
  {
    long n = (long)NA * KA;
    cvt_w1<<<dim3((unsigned)((n + 255) / 256)), 256, 0, stream>>>(w1, b1, W1t);
  }
  {
    long n = (long)OUT_F * KB;
    cvt_w2<<<dim3((unsigned)((n + 255) / 256)), 256, 0, stream>>>(w2, b2, W2t);
  }
  routing<<<BATCH / 8, 256, 0, stream>>>(x, nw, nb, mix, Hs);

  dim3 gA(NA / 128, BATCH / 128);     // 32 x 64 blocks
  gemm_wmma<0><<<gA, 256, 0, stream>>>(Xb, W1t, KA, nullptr, Hs, mix);

  dim3 gB(OUT_F / 128, BATCH / 128);  // 8 x 64 blocks
  gemm_wmma<1><<<gB, 256, 0, stream>>>(Hs, W2t, KB, out, nullptr, nullptr);
}
